// MLAttention_32177894981776
// MI455X (gfx1250) — matrix operand order repaired and checked by reading, not compiled, not run on hardware
//
#include <hip/hip_runtime.h>
#include <math.h>

// ---------------------------------------------------------------------------
// MLA forward for MI455X (gfx1250, wave32, WMMA bf16 16x16x32)
// ---------------------------------------------------------------------------

typedef __bf16 bf16;
typedef __bf16 bf16x8  __attribute__((ext_vector_type(8)));
typedef __bf16 bf16x16 __attribute__((ext_vector_type(16)));
typedef float  f32x4   __attribute__((ext_vector_type(4)));
typedef float  f32x8   __attribute__((ext_vector_type(8)));
typedef int    i32x4   __attribute__((ext_vector_type(4)));

// ---- gfx1250 async global->LDS path (guarded; falls back to sync loads) ----
#if defined(__has_builtin)
#  if __has_builtin(__builtin_amdgcn_global_load_async_to_lds_b128) && \
      __has_builtin(__builtin_amdgcn_s_wait_asynccnt)
#    define HAVE_ASYNC_LDS 1
#  endif
#endif
#ifndef HAVE_ASYNC_LDS
#  define HAVE_ASYNC_LDS 0
#endif

#if HAVE_ASYNC_LDS
typedef __attribute__((address_space(1))) i32x4 gas_i32x4;  // global b128
typedef __attribute__((address_space(3))) i32x4 las_i32x4;  // LDS b128
// Copy 16 bytes global->LDS without a VGPR round trip (ASYNCcnt-tracked).
// Low 32 bits of a flat shared address are the LDS byte offset (ISA 10.2).
__device__ __forceinline__ void async_copy16(void* lds, const void* g) {
  __builtin_amdgcn_global_load_async_to_lds_b128(
      (gas_i32x4*)(size_t)g, (las_i32x4*)(unsigned)(size_t)lds, 0, 0);
}
#endif

__device__ __forceinline__ bf16x16 frag_cat(bf16x8 lo, bf16x8 hi) {
  return __builtin_shufflevector(lo, hi, 0,1,2,3,4,5,6,7,8,9,10,11,12,13,14,15);
}

__device__ __forceinline__ f32x8 wmma_bf16(bf16x16 a, bf16x16 b, f32x8 c) {
  // (neg_a, A, neg_b, B, c_mod, C, reuse_a, reuse_b)
  return __builtin_amdgcn_wmma_f32_16x16x32_bf16(false, a, false, b, (short)0, c,
                                                 false, false);
}

// ---------------------------------------------------------------------------
// fp32 -> bf16 conversion, 8 elements per thread (HBM-rate)
// ---------------------------------------------------------------------------
__global__ void cvt_bf16_kernel(const float* __restrict__ in,
                                bf16* __restrict__ out, size_t n8) {
  size_t i = (size_t)blockIdx.x * blockDim.x + threadIdx.x;
  if (i >= n8) return;
  const float* p = in + i * 8;
  f32x4 a = *(const f32x4*)p;
  f32x4 b = *(const f32x4*)(p + 4);
  bf16x8 o;
#pragma unroll
  for (int t = 0; t < 4; t++) { o[t] = (bf16)a[t]; o[t + 4] = (bf16)b[t]; }
  *(bf16x8*)(out + i * 8) = o;
}

// ---------------------------------------------------------------------------
// Tiled GEMM:  C[M,N] (f32) = A[M,K] (bf16, row major) * W[N,K]^T (bf16)
// Block tile 128x128, k-step 32, 8 waves (4x2), each wave 32x64 (2x4 WMMA).
// Double-buffered async global->LDS tile staging on gfx1250.
// ---------------------------------------------------------------------------
__global__ __launch_bounds__(256)
void gemm_bf16_kernel(const bf16* __restrict__ A, const bf16* __restrict__ W,
                      float* __restrict__ C, int M, int N, int K) {
  __shared__ bf16 lA[2][128 * 40];   // stride 40 (pad) to spread LDS banks
  __shared__ bf16 lB[2][128 * 40];

  const int tid  = threadIdx.x;
  const int lane = tid & 31;
  const int wave = tid >> 5;
  const int wm   = wave >> 1;            // 0..3 -> 32-row slice
  const int wn   = wave & 1;             // 0..1 -> 64-col slice
  const int m0   = blockIdx.y * 128;
  const int n0   = blockIdx.x * 128;

  const int lr     = lane & 15;
  const int rh     = (lane >> 4) * 8;
  const int koff8  = (lane < 16) ? 0 : 8;    // A-fragment K offset
  const int koff16 = (lane < 16) ? 0 : 16;   // unused; B fragments also use koff8

  // cooperative tile staging: 256 threads, each owns 16 bf16 of one row
  const int lrow = tid >> 1;             // 0..127
  const int lcol = (tid & 1) * 16;       // 0 or 16
  const int arow = m0 + lrow;
  int wrow = n0 + lrow;
  if (wrow > N - 1) wrow = N - 1;        // guard for N not multiple of 128

  f32x8 acc[2][4] = {};
  const int NT = K >> 5;                 // number of 32-wide k tiles

#if HAVE_ASYNC_LDS
  {
    const bf16* ag = A + (size_t)arow * K + lcol;
    const bf16* wg = W + (size_t)wrow * K + lcol;
    async_copy16(&lA[0][lrow * 40 + lcol],     ag);
    async_copy16(&lA[0][lrow * 40 + lcol + 8], ag + 8);
    async_copy16(&lB[0][lrow * 40 + lcol],     wg);
    async_copy16(&lB[0][lrow * 40 + lcol + 8], wg + 8);
  }
#endif

  for (int it = 0; it < NT; ++it) {
#if HAVE_ASYNC_LDS
    const int cb = it & 1;
    if (it + 1 < NT) {
      // kick off next tile's DMA into the other buffer, then wait for the
      // current tile only (async loads complete in order; 4 remain in flight)
      const int k1 = (it + 1) << 5;
      const bf16* ag = A + (size_t)arow * K + k1 + lcol;
      const bf16* wg = W + (size_t)wrow * K + k1 + lcol;
      async_copy16(&lA[cb ^ 1][lrow * 40 + lcol],     ag);
      async_copy16(&lA[cb ^ 1][lrow * 40 + lcol + 8], ag + 8);
      async_copy16(&lB[cb ^ 1][lrow * 40 + lcol],     wg);
      async_copy16(&lB[cb ^ 1][lrow * 40 + lcol + 8], wg + 8);
      __builtin_amdgcn_s_wait_asynccnt(4);
    } else {
      __builtin_amdgcn_s_wait_asynccnt(0);
    }
    __syncthreads();                    // publish current tile to all waves
#else
    const int cb = 0;
    const int k0 = it << 5;
    __syncthreads();
    const bf16* ag = A + (size_t)arow * K + k0 + lcol;
    const bf16* wg = W + (size_t)wrow * K + k0 + lcol;
    bf16x8 a0 = *(const bf16x8*)(ag);
    bf16x8 a1 = *(const bf16x8*)(ag + 8);
    bf16x8 b0 = *(const bf16x8*)(wg);
    bf16x8 b1 = *(const bf16x8*)(wg + 8);
    *(bf16x8*)(&lA[0][lrow * 40 + lcol])     = a0;
    *(bf16x8*)(&lA[0][lrow * 40 + lcol + 8]) = a1;
    *(bf16x8*)(&lB[0][lrow * 40 + lcol])     = b0;
    *(bf16x8*)(&lB[0][lrow * 40 + lcol + 8]) = b1;
    if (k0 + 32 < K) {
      __builtin_prefetch(ag + 32, 0, 3);
      __builtin_prefetch(wg + 32, 0, 3);
    }
    __syncthreads();
#endif

    bf16x16 af[2];
#pragma unroll
    for (int i = 0; i < 2; i++) {
      const bf16* p = &lA[cb][(wm * 32 + i * 16 + lr) * 40 + koff8];
      af[i] = frag_cat(*(const bf16x8*)p, *(const bf16x8*)(p + 16));
    }
    bf16x16 bfv[4];
#pragma unroll
    for (int j = 0; j < 4; j++) {
      const bf16* p = &lB[cb][(wn * 64 + j * 16 + lr) * 40 + koff8];
      bfv[j] = frag_cat(*(const bf16x8*)p, *(const bf16x8*)(p + 16));
    }
#pragma unroll
    for (int i = 0; i < 2; i++)
#pragma unroll
      for (int j = 0; j < 4; j++)
        acc[i][j] = wmma_bf16(af[i], bfv[j], acc[i][j]);

#if HAVE_ASYNC_LDS
    __syncthreads();                    // all waves done reading buffer cb
#endif
  }

#pragma unroll
  for (int i = 0; i < 2; i++)
#pragma unroll
    for (int j = 0; j < 4; j++) {
      int col = n0 + wn * 64 + j * 16 + lr;
      if (col < N) {
#pragma unroll
        for (int r = 0; r < 8; r++) {
          int row = m0 + wm * 32 + i * 16 + r + rh;
          C[(size_t)row * N + col] = acc[i][j][r];
        }
      }
    }
}

// ---------------------------------------------------------------------------
// RMSNorm (fp32 in, bf16 out). One block per row.
// ---------------------------------------------------------------------------
__global__ __launch_bounds__(256)
void rmsnorm_bf16_kernel(const float* __restrict__ X, const float* __restrict__ w,
                         bf16* __restrict__ Y, int C, int in_stride) {
  __shared__ float red[256];
  const int row = blockIdx.x;
  const float* x = X + (size_t)row * in_stride;
  float s = 0.f;
  for (int c = threadIdx.x; c < C; c += 256) { float v = x[c]; s += v * v; }
  red[threadIdx.x] = s;
  __syncthreads();
  for (int o2 = 128; o2 > 0; o2 >>= 1) {
    if (threadIdx.x < o2) red[threadIdx.x] += red[threadIdx.x + o2];
    __syncthreads();
  }
  const float rms = rsqrtf(red[0] / (float)C + 1e-6f);
  bf16* y = Y + (size_t)row * C;
  for (int c = threadIdx.x; c < C; c += 256) y[c] = (bf16)(x[c] * rms * w[c]);
}

// ---------------------------------------------------------------------------
// RoPE on the shared k_rope slice of kv_raw (cols 512..575) -> bf16 [BT,64]
// ---------------------------------------------------------------------------
__global__ void rope_k_kernel(const float* __restrict__ kv_raw,
                              const float* __restrict__ cosT,
                              const float* __restrict__ sinT,
                              bf16* __restrict__ kr, int T, int nrows) {
  int idx = blockIdx.x * blockDim.x + threadIdx.x;
  int row = idx >> 5, i = idx & 31;
  if (row >= nrows) return;
  int t = row % T;
  const float* src = kv_raw + (size_t)row * 576 + 512 + 2 * i;
  float x0 = src[0], x1 = src[1];
  float c = cosT[t * 32 + i], s = sinT[t * 32 + i];
  kr[(size_t)row * 64 + 2 * i]     = (bf16)(x0 * c - x1 * s);
  kr[(size_t)row * 64 + 2 * i + 1] = (bf16)(x0 * s + x1 * c);
}

// ---------------------------------------------------------------------------
// Pack kernels: build q [B,H,T,192], k [B,H,T,192], vT [B,H,128,T] in bf16
// ---------------------------------------------------------------------------
__global__ void pack_q_nope_kernel(const float* __restrict__ qn,
                                   bf16* __restrict__ qh, int T) {
  int idx = blockIdx.x * blockDim.x + threadIdx.x;   // B*T*H*128 threads
  int d = idx & 127, rest = idx >> 7;
  int h = rest & 15, bt = rest >> 4;
  int t = bt % T, b = bt / T;
  qh[(((size_t)(b * 16 + h) * T + t) * 192) + d] =
      (bf16)qn[(size_t)bt * 2048 + h * 128 + d];
}

__global__ void pack_q_rope_kernel(const float* __restrict__ qr,
                                   const float* __restrict__ cosT,
                                   const float* __restrict__ sinT,
                                   bf16* __restrict__ qh, int T) {
  int idx = blockIdx.x * blockDim.x + threadIdx.x;   // B*T*H*32 threads
  int i = idx & 31, rest = idx >> 5;
  int h = rest & 15, bt = rest >> 4;
  int t = bt % T, b = bt / T;
  float x0 = qr[(size_t)bt * 1024 + h * 64 + 2 * i];
  float x1 = qr[(size_t)bt * 1024 + h * 64 + 2 * i + 1];
  float c = cosT[t * 32 + i], s = sinT[t * 32 + i];
  bf16* dst = qh + (((size_t)(b * 16 + h) * T + t) * 192) + 128 + 2 * i;
  dst[0] = (bf16)(x0 * c - x1 * s);
  dst[1] = (bf16)(x0 * s + x1 * c);
}

__global__ void pack_k_kernel(const float* __restrict__ kvup,
                              const bf16* __restrict__ kr,
                              bf16* __restrict__ kh, int T) {
  int idx = blockIdx.x * blockDim.x + threadIdx.x;   // B*T*H*192 threads
  int d = idx % 192, rest = idx / 192;
  int h = rest & 15, bt = rest >> 4;
  int t = bt % T, b = bt / T;
  bf16 v;
  if (d < 128) v = (bf16)kvup[(size_t)bt * 4096 + h * 256 + d];
  else         v = kr[(size_t)bt * 64 + (d - 128)];
  kh[(((size_t)(b * 16 + h) * T + t) * 192) + d] = v;
}

__global__ void pack_vT_kernel(const float* __restrict__ kvup,
                               bf16* __restrict__ vT, int T) {
  int idx = blockIdx.x * blockDim.x + threadIdx.x;   // B*T*H*128 threads
  int d = idx & 127, rest = idx >> 7;
  int h = rest & 15, bt = rest >> 4;
  int t = bt % T, b = bt / T;
  vT[(((size_t)(b * 16 + h) * 128 + d) * T) + t] =
      (bf16)kvup[(size_t)bt * 4096 + h * 256 + 128 + d];
}

// ---------------------------------------------------------------------------
// Flash attention (causal), bf16 WMMA, online softmax.
// Grid (T/128, H, B); 8 waves/block, each wave owns 16 query rows.
// q,k: [B,H,T,192]  vT: [B,H,128,T]  out: [B,T,H,128] bf16
// K/V stream from global (per-(b,h) working set is 1.25 MB -> L2 resident).
// ---------------------------------------------------------------------------
__global__ __launch_bounds__(256)
void flash_attn_kernel(const bf16* __restrict__ Q, const bf16* __restrict__ Kh,
                       const bf16* __restrict__ VT, bf16* __restrict__ O,
                       int T) {
  __shared__ bf16 lP[8][16 * 136];     // per-wave P pane, stride 136 (16B ok)

  const int lane = threadIdx.x & 31;
  const int wave = threadIdx.x >> 5;
  const int qblk = blockIdx.x;
  const int h = blockIdx.y, b = blockIdx.z;
  const int bh = b * 16 + h;
  const int lr = lane & 15;
  const int rh = (lane >> 4) * 8;
  const int koff8  = (lane < 16) ? 0 : 8;
  const int koff16 = (lane < 16) ? 0 : 16;
  const int qrow = qblk * 128 + wave * 16;
  const float scale = 0.072168784f;    // 1/sqrt(192)

  // Q A-fragments for this wave's 16 rows (K=192 -> 6 wmma k-steps)
  bf16x16 aq[6];
  const bf16* qbase = Q + ((size_t)bh * T + qrow + lr) * 192;
#pragma unroll
  for (int kk = 0; kk < 6; kk++) {
    const bf16* p = qbase + kk * 32 + koff8;
    aq[kk] = frag_cat(*(const bf16x8*)p, *(const bf16x8*)(p + 16));
  }

  f32x8 o[8] = {};
  float m_r[8], l_r[8];
#pragma unroll
  for (int r = 0; r < 8; r++) { m_r[r] = -1e30f; l_r[r] = 0.f; }

  for (int sblk = 0; sblk <= qblk; sblk++) {
    // ---- S = Q K^T on a 16x128 stripe ----
    f32x8 sj[8];
#pragma unroll
    for (int j = 0; j < 8; j++) {
      f32x8 s = {};
      const bf16* kb =
          Kh + ((size_t)bh * T + sblk * 128 + j * 16 + lr) * 192 + koff8;
#pragma unroll
      for (int kk = 0; kk < 6; kk++) {
        const bf16* p = kb + kk * 32;
        bf16x16 bk = frag_cat(*(const bf16x8*)p, *(const bf16x8*)(p + 16));
        s = wmma_bf16(aq[kk], bk, s);
      }
      sj[j] = s;
    }

    // ---- scale + causal mask (only diagonal block needs masking) ----
    const bool diag = (sblk == qblk);
#pragma unroll
    for (int j = 0; j < 8; j++) {
      int col = sblk * 128 + j * 16 + lr;
#pragma unroll
      for (int r = 0; r < 8; r++) {
        float v = sj[j][r] * scale;
        if (diag && col > qrow + r + rh) v = -1e9f;
        sj[j][r] = v;
      }
    }

    // ---- row max (across 8 col-tiles, then across 16 lanes) ----
    float mx[8];
#pragma unroll
    for (int r = 0; r < 8; r++) {
      float m = sj[0][r];
#pragma unroll
      for (int j = 1; j < 8; j++) m = fmaxf(m, sj[j][r]);
      mx[r] = m;
    }
#pragma unroll
    for (int off = 1; off < 16; off <<= 1)
#pragma unroll
      for (int r = 0; r < 8; r++)
        mx[r] = fmaxf(mx[r], __shfl_xor(mx[r], off, 32));

    // ---- online softmax state ----
    float alpha[8];
#pragma unroll
    for (int r = 0; r < 8; r++) {
      float mn = fmaxf(m_r[r], mx[r]);
      alpha[r] = __expf(m_r[r] - mn);
      m_r[r] = mn;
    }
    float rs[8];
#pragma unroll
    for (int r = 0; r < 8; r++) {
      float srow = 0.f;
#pragma unroll
      for (int j = 0; j < 8; j++) {
        float p = __expf(sj[j][r] - m_r[r]);
        sj[j][r] = p;
        srow += p;
      }
      rs[r] = srow;
    }
#pragma unroll
    for (int off = 1; off < 16; off <<= 1)
#pragma unroll
      for (int r = 0; r < 8; r++) rs[r] += __shfl_xor(rs[r], off, 32);
#pragma unroll
    for (int r = 0; r < 8; r++) l_r[r] = l_r[r] * alpha[r] + rs[r];
#pragma unroll
    for (int j = 0; j < 8; j++)
#pragma unroll
      for (int r = 0; r < 8; r++) o[j][r] *= alpha[r];

    // ---- relayout P (C-layout) -> A-layout through LDS ----
    bf16* pw = &lP[wave][0];
#pragma unroll
    for (int j = 0; j < 8; j++)
#pragma unroll
      for (int r = 0; r < 8; r++)
        pw[(r + rh) * 136 + j * 16 + lr] = (bf16)sj[j][r];
    __syncthreads();

    // ---- O += P V  (K=128 -> 4 wmma k-steps, N=128 -> 8 tiles) ----
#pragma unroll
    for (int kk = 0; kk < 4; kk++) {
      const bf16* pa = &lP[wave][lr * 136 + kk * 32 + koff8];
      bf16x16 ap = frag_cat(*(const bf16x8*)pa, *(const bf16x8*)(pa + 16));
#pragma unroll
      for (int j = 0; j < 8; j++) {
        const bf16* vb = VT + ((size_t)bh * 128 + j * 16 + lr) * T +
                         sblk * 128 + kk * 32 + koff8;
        bf16x16 bv = frag_cat(*(const bf16x8*)vb, *(const bf16x8*)(vb + 16));
        o[j] = wmma_bf16(ap, bv, o[j]);
      }
    }
    __syncthreads();
  }

  // ---- finalize and store [B,T,H,128] bf16 ----
#pragma unroll
  for (int j = 0; j < 8; j++) {
    int d = j * 16 + lr;
#pragma unroll
    for (int r = 0; r < 8; r++) {
      int t = qrow + r + rh;
      O[(((size_t)b * T + t) * 16 + h) * 128 + d] = (bf16)(o[j][r] / l_r[r]);
    }
  }
}

// ---------------------------------------------------------------------------
// Host-side launch
// ---------------------------------------------------------------------------
extern "C" void kernel_launch(void* const* d_in, const int* in_sizes, int n_in,
                              void* d_out, int out_size, void* d_ws,
                              size_t ws_size, hipStream_t stream) {
  (void)in_sizes; (void)n_in; (void)out_size; (void)ws_size;
  enum { B_ = 4, T_ = 2048, DIM_ = 2048, H_ = 16, BT_ = B_ * T_,
         QL_ = 1536, KVL_ = 512, RD_ = 64, ND_ = 128, VD_ = 128 };

  const float* x     = (const float*)d_in[0];
  const float* fcos  = (const float*)d_in[1];
  const float* fsin  = (const float*)d_in[2];
  /* d_in[3] = mask: causal mask is applied analytically */
  const float* w_qd  = (const float*)d_in[4];
  const float* w_qn  = (const float*)d_in[5];   // q_norm
  const float* w_qun = (const float*)d_in[6];
  const float* w_qur = (const float*)d_in[7];
  const float* w_kd  = (const float*)d_in[8];
  const float* w_kn  = (const float*)d_in[9];   // kv_norm
  const float* w_ku  = (const float*)d_in[10];
  const float* w_wo  = (const float*)d_in[11];
  float* out = (float*)d_out;

  char* ws = (char*)d_ws;
  size_t off = 0;
  auto alloc = [&](size_t bytes) -> void* {
    void* p = ws + off;
    off = (off + bytes + 255) & ~(size_t)255;
    return p;
  };

  bf16* xb   = (bf16*)alloc((size_t)BT_ * DIM_ * 2);
  bf16* wqd  = (bf16*)alloc((size_t)QL_ * DIM_ * 2);
  bf16* wqun = (bf16*)alloc((size_t)(H_ * ND_) * QL_ * 2);
  bf16* wqur = (bf16*)alloc((size_t)(H_ * RD_) * QL_ * 2);
  bf16* wkd  = (bf16*)alloc((size_t)(KVL_ + RD_) * DIM_ * 2);
  bf16* wku  = (bf16*)alloc((size_t)(H_ * (ND_ + VD_)) * KVL_ * 2);
  bf16* wwo  = (bf16*)alloc((size_t)DIM_ * (H_ * VD_) * 2);
  float* cq_raw = (float*)alloc((size_t)BT_ * QL_ * 4);
  bf16*  cq     = (bf16*)alloc((size_t)BT_ * QL_ * 2);
  float* kv_raw = (float*)alloc((size_t)BT_ * (KVL_ + RD_) * 4);
  bf16*  ckv    = (bf16*)alloc((size_t)BT_ * KVL_ * 2);
  bf16*  krope  = (bf16*)alloc((size_t)BT_ * RD_ * 2);
  float* qnope  = (float*)alloc((size_t)BT_ * (H_ * ND_) * 4);
  float* qrope  = (float*)alloc((size_t)BT_ * (H_ * RD_) * 4);
  float* kvup   = (float*)alloc((size_t)BT_ * (H_ * (ND_ + VD_)) * 4);
  bf16*  qh     = (bf16*)alloc((size_t)B_ * H_ * T_ * 192 * 2);
  bf16*  kh     = (bf16*)alloc((size_t)B_ * H_ * T_ * 192 * 2);
  bf16*  vT     = (bf16*)alloc((size_t)B_ * H_ * 128 * T_ * 2);
  bf16*  attnb  = (bf16*)alloc((size_t)BT_ * (H_ * VD_) * 2);

  auto cvt = [&](const float* src, bf16* dst, size_t n) {
    size_t n8 = n / 8;
    cvt_bf16_kernel<<<(unsigned)((n8 + 255) / 256), 256, 0, stream>>>(src, dst,
                                                                      n8);
  };
  auto gemm = [&](const bf16* A, const bf16* W, float* C, int M, int N, int K) {
    dim3 grid((N + 127) / 128, M / 128);
    gemm_bf16_kernel<<<grid, 256, 0, stream>>>(A, W, C, M, N, K);
  };

  // 1) bf16 conversions
  cvt(x, xb, (size_t)BT_ * DIM_);
  cvt(w_qd, wqd, (size_t)QL_ * DIM_);
  cvt(w_qun, wqun, (size_t)(H_ * ND_) * QL_);
  cvt(w_qur, wqur, (size_t)(H_ * RD_) * QL_);
  cvt(w_kd, wkd, (size_t)(KVL_ + RD_) * DIM_);
  cvt(w_ku, wku, (size_t)(H_ * (ND_ + VD_)) * KVL_);
  cvt(w_wo, wwo, (size_t)DIM_ * (H_ * VD_));

  // 2) down-projections
  gemm(xb, wqd, cq_raw, BT_, QL_, DIM_);
  gemm(xb, wkd, kv_raw, BT_, KVL_ + RD_, DIM_);

  // 3) rmsnorm + k-rope
  rmsnorm_bf16_kernel<<<BT_, 256, 0, stream>>>(cq_raw, w_qn, cq, QL_, QL_);
  rmsnorm_bf16_kernel<<<BT_, 256, 0, stream>>>(kv_raw, w_kn, ckv, KVL_,
                                               KVL_ + RD_);
  rope_k_kernel<<<(BT_ * 32) / 256, 256, 0, stream>>>(kv_raw, fcos, fsin, krope,
                                                      T_, BT_);

  // 4) up-projections
  gemm(cq, wqun, qnope, BT_, H_ * ND_, QL_);
  gemm(cq, wqur, qrope, BT_, H_ * RD_, QL_);
  gemm(ckv, wku, kvup, BT_, H_ * (ND_ + VD_), KVL_);

  // 5) pack q/k/v into attention layouts
  pack_q_nope_kernel<<<(BT_ * H_ * 128) / 256, 256, 0, stream>>>(qnope, qh, T_);
  pack_q_rope_kernel<<<(BT_ * H_ * 32) / 256, 256, 0, stream>>>(qrope, fcos,
                                                                fsin, qh, T_);
  pack_k_kernel<<<(BT_ * H_ * 192) / 256, 256, 0, stream>>>(kvup, krope, kh, T_);
  pack_vT_kernel<<<(BT_ * H_ * 128) / 256, 256, 0, stream>>>(kvup, vT, T_);

  // 6) flash attention
  {
    dim3 grid(T_ / 128, H_, B_);
    flash_attn_kernel<<<grid, 256, 0, stream>>>(qh, kh, vT, attnb, T_);
  }

  // 7) output projection -> d_out (fp32)
  gemm(attnb, wwo, out, BT_, DIM_, H_ * VD_);
}
